// Attention_76974403879144
// MI455X (gfx1250) — matrix operand order repaired and checked by reading, not compiled, not run on hardware
//
#include <hip/hip_runtime.h>

#define B_   256
#define N_   196
#define ENC_ 1792
#define DEC_ 512
#define ATT_ 512

typedef __attribute__((ext_vector_type(16))) _Float16 v16h;
typedef __attribute__((ext_vector_type(8)))  float    v8f;

// pack two f32 -> packed 2xf16 dword (v_cvt_f16_f32 x2 + v_pack_b32_f16)
__device__ __forceinline__ unsigned pk2h(float a, float b) {
  union { _Float16 h[2]; unsigned u; } p;
  p.h[0] = (_Float16)a;
  p.h[1] = (_Float16)b;
  return p.u;
}

// ---------------------------------------------------------------------------
// Kernel 1: att2[b,c] = decoder_hidden[b,:] @ W_dec[:,c] + b_dec[c]
// ---------------------------------------------------------------------------
__global__ void att2_kernel(const float* __restrict__ dh,
                            const float* __restrict__ Wd,
                            const float* __restrict__ bd,
                            float* __restrict__ att2) {
  __shared__ float sdh[DEC_];
  const int b = blockIdx.x, t = threadIdx.x;
  sdh[t]       = dh[b * DEC_ + t];
  sdh[t + 256] = dh[b * DEC_ + t + 256];
  __syncthreads();
  #pragma unroll
  for (int h = 0; h < 2; ++h) {
    const int c = t + h * 256;
    float acc = bd[c];
    #pragma unroll 8
    for (int k = 0; k < DEC_; ++k)
      acc = fmaf(sdh[k], Wd[k * ATT_ + c], acc);   // coalesced across threads
    att2[b * ATT_ + c] = acc;
  }
}

// ---------------------------------------------------------------------------
// Kernel 2: fused  att[m] = relu(enc[m,:]@W_enc + b_enc + att2[b,:]) . W_full + b_full
// WG = 256 threads = 8 waves: 4 M-tiles (16 rows) x 2 column halves (256 cols).
// f32 HBM -> packed-f16 LDS staging, v_wmma_f32_16x16x32_f16, f32 accumulate,
// epilogue fused (relu + W_full dot is separable across column tiles).
// ---------------------------------------------------------------------------
__global__ __launch_bounds__(256)
void score_kernel(const float* __restrict__ enc,
                  const float* __restrict__ We,
                  const float* __restrict__ be,
                  const float* __restrict__ Wf,
                  const float* __restrict__ bf,
                  const float* __restrict__ att2,
                  float* __restrict__ att) {
  // B tile transposed [col][k], +2 pad => 17-bank stride, conflict-free frag reads
  __shared__ __attribute__((aligned(16))) _Float16 lBt[ATT_][34];
  __shared__ __attribute__((aligned(16))) _Float16 lA[4][16][34];
  __shared__ float sScore[64];

  const int tid  = threadIdx.x;
  const int lane = tid & 31;
  const int wid  = tid >> 5;
  const int mt   = wid & 3;      // M-tile within workgroup
  const int ch   = wid >> 2;     // column half: 0 -> cols 0..255, 1 -> 256..511
  const int hi   = lane >> 4;    // lane half (wave32 WMMA layout)
  const int ln   = lane & 15;
  const int rowBase = blockIdx.x * 64;   // 50176 / 64 = 784 blocks, exact

  if (tid < 64) sScore[tid] = 0.0f;

  v8f acc[16];
  #pragma unroll
  for (int i = 0; i < 16; ++i) { v8f z = {}; acc[i] = z; }

  const float4* enc4 = (const float4*)enc;
  const float4* We4  = (const float4*)We;

  for (int k0 = 0; k0 < ENC_; k0 += 32) {          // 56 K-steps
    __syncthreads();   // protect LDS from previous iteration's readers
    // ---- stage B: W_enc[k0..k0+31][0..511] -> lBt (packed f16, transposed) ----
    // Each thread: 2 float4 loads (two K rows, same 4 cols) -> 4 packed b32 stores.
    #pragma unroll
    for (int i = 0; i < 8; ++i) {
      const int idx = i * 256 + tid;               // 0..2047
      const int rp  = idx >> 7;                    // K pair 0..15
      const int c4  = idx & 127;                   // float4 column
      const float4 w0 = We4[(k0 + 2 * rp)     * (ATT_ / 4) + c4];  // coalesced
      const float4 w1 = We4[(k0 + 2 * rp + 1) * (ATT_ / 4) + c4];  // coalesced
      *(unsigned*)&lBt[c4 * 4 + 0][2 * rp] = pk2h(w0.x, w1.x);
      *(unsigned*)&lBt[c4 * 4 + 1][2 * rp] = pk2h(w0.y, w1.y);
      *(unsigned*)&lBt[c4 * 4 + 2][2 * rp] = pk2h(w0.z, w1.z);
      *(unsigned*)&lBt[c4 * 4 + 3][2 * rp] = pk2h(w0.w, w1.w);
    }
    // ---- stage A: enc rows rowBase..rowBase+63, K slice of 32 ----
    // One float4 = 4 consecutive K of one row -> 2 packed b32 stores.
    #pragma unroll
    for (int i = 0; i < 2; ++i) {
      const int idx = i * 256 + tid;               // 0..511
      const int r   = idx >> 3;                    // row 0..63
      const int c4  = idx & 7;                     // float4 within K-slice
      const float4 w = enc4[(size_t)(rowBase + r) * (ENC_ / 4) + (k0 >> 2) + c4];
      *(unsigned*)&lA[r >> 4][r & 15][c4 * 4 + 0] = pk2h(w.x, w.y);
      *(unsigned*)&lA[r >> 4][r & 15][c4 * 4 + 2] = pk2h(w.z, w.w);
    }
    __syncthreads();

    // ---- A fragment: ISA 16-bit 16x32 layout (lane m=ln, halves by hi) ----
    union { unsigned u[8]; v16h v; } afr;
    #pragma unroll
    for (int v = 0; v < 8; ++v) {
      const int kb = (v >> 2) * 16 + hi * 8 + (v & 3) * 2;   // even -> b32 load
      afr.u[v] = *(const unsigned*)&lA[mt][ln][kb];
    }
    // ---- 16 column tiles: B frag gather + WMMA ----
    #pragma unroll
    for (int nt = 0; nt < 16; ++nt) {
      union { unsigned u[8]; v16h v; } bfr;
      const int col = ch * 256 + nt * 16 + ln;
      #pragma unroll
      for (int v = 0; v < 8; ++v) {
        const int kb = (v >> 2) * 16 + hi * 8 + (v & 3) * 2;   // K pair per VGPR
        bfr.u[v] = *(const unsigned*)&lBt[col][kb];
      }
      acc[nt] = __builtin_amdgcn_wmma_f32_16x16x32_f16(
          false, afr.v, false, bfr.v, (short)0, acc[nt], false, false);
    }
  }

  // ---- fused epilogue: + b_enc + att2[b], relu, dot with W_full ----
  float part[8];
  #pragma unroll
  for (int j = 0; j < 8; ++j) part[j] = 0.0f;
  #pragma unroll
  for (int nt = 0; nt < 16; ++nt) {
    const int col = ch * 256 + nt * 16 + ln;       // D layout: N = lane%16
    const float wf  = Wf[col];
    const float bec = be[col];
    #pragma unroll
    for (int j = 0; j < 8; ++j) {                  // D layout: M = j + 8*hi
      const int row = rowBase + mt * 16 + hi * 8 + j;
      const int b   = row / N_;
      float x = acc[nt][j] + bec + att2[b * ATT_ + col];
      x = fmaxf(x, 0.0f);
      part[j] = fmaf(x, wf, part[j]);
    }
  }
  // reduce over the 16 lanes of each half (xor 1,2,4,8 stays within halves)
  #pragma unroll
  for (int j = 0; j < 8; ++j) {
    #pragma unroll
    for (int off = 1; off < 16; off <<= 1)
      part[j] += __shfl_xor(part[j], off, 32);
  }
  if (ln == 0) {
    #pragma unroll
    for (int j = 0; j < 8; ++j)
      atomicAdd(&sScore[mt * 16 + hi * 8 + j], part[j]);   // ds_add_f32
  }
  __syncthreads();
  if (tid < 64) att[rowBase + tid] = sScore[tid] + bf[0];
}

// ---------------------------------------------------------------------------
// Kernel 3: softmax over N=196 per batch
// ---------------------------------------------------------------------------
__global__ void softmax_kernel(const float* __restrict__ att,
                               float* __restrict__ alpha) {
  __shared__ float red[256];
  const int b = blockIdx.x, t = threadIdx.x;
  const float v = (t < N_) ? att[b * N_ + t] : -1e30f;
  red[t] = v;
  __syncthreads();
  for (int s = 128; s > 0; s >>= 1) {
    if (t < s) red[t] = fmaxf(red[t], red[t + s]);
    __syncthreads();
  }
  const float m = red[0];
  __syncthreads();
  const float e = (t < N_) ? __expf(v - m) : 0.0f;
  red[t] = e;
  __syncthreads();
  for (int s = 128; s > 0; s >>= 1) {
    if (t < s) red[t] += red[t + s];
    __syncthreads();
  }
  if (t < N_) alpha[b * N_ + t] = e * (1.0f / red[0]);
}

// ---------------------------------------------------------------------------
// Kernel 4: awe[b,e] = sum_n alpha[b,n] * enc[b,n,e]   (HBM-bound float4 stream)
// ---------------------------------------------------------------------------
__global__ void awe_kernel(const float* __restrict__ enc,
                           const float* __restrict__ alpha,
                           float* __restrict__ awe) {
  __shared__ float sA[N_];
  const int b = blockIdx.y, t = threadIdx.x;
  if (t < N_) sA[t] = alpha[b * N_ + t];
  __syncthreads();
  const int c4 = blockIdx.x * 224 + t;             // 0..447 (ENC/4 = 448 = 2*224)
  const float4* enc4 = (const float4*)enc;
  float4 acc = {0.0f, 0.0f, 0.0f, 0.0f};
  for (int n = 0; n < N_; ++n) {
    const float a = sA[n];
    const float4 e = enc4[(size_t)(b * N_ + n) * (ENC_ / 4) + c4];
    acc.x = fmaf(a, e.x, acc.x);
    acc.y = fmaf(a, e.y, acc.y);
    acc.z = fmaf(a, e.z, acc.z);
    acc.w = fmaf(a, e.w, acc.w);
  }
  ((float4*)awe)[(size_t)b * (ENC_ / 4) + c4] = acc;
}

// ---------------------------------------------------------------------------
extern "C" void kernel_launch(void* const* d_in, const int* in_sizes, int n_in,
                              void* d_out, int out_size, void* d_ws, size_t ws_size,
                              hipStream_t stream) {
  (void)in_sizes; (void)n_in; (void)out_size; (void)ws_size;
  const float* enc = (const float*)d_in[0];   // [B,N,ENC]
  const float* dh  = (const float*)d_in[1];   // [B,DEC]
  const float* We  = (const float*)d_in[2];   // [ENC,ATT]
  const float* be  = (const float*)d_in[3];   // [ATT]
  const float* Wd  = (const float*)d_in[4];   // [DEC,ATT]
  const float* bd  = (const float*)d_in[5];   // [ATT]
  const float* Wf  = (const float*)d_in[6];   // [ATT]
  const float* bf  = (const float*)d_in[7];   // scalar

  float* awe_out   = (float*)d_out;                         // [B,ENC] first
  float* alpha_out = awe_out + (size_t)B_ * ENC_;           // [B,N] second
  float* att2_ws   = (float*)d_ws;                          // [B,ATT]
  float* att_ws    = att2_ws + (size_t)B_ * ATT_;           // [B,N]

  att2_kernel   <<<B_, 256, 0, stream>>>(dh, Wd, bd, att2_ws);
  score_kernel  <<<(B_ * N_) / 64, 256, 0, stream>>>(enc, We, be, Wf, bf, att2_ws, att_ws);
  softmax_kernel<<<B_, 256, 0, stream>>>(att_ws, alpha_out);
  awe_kernel    <<<dim3(2, B_), 224, 0, stream>>>(enc, alpha_out, awe_out);
}
